// Encoder_34815004901450
// MI455X (gfx1250) — matrix operand order repaired and checked by reading, not compiled, not run on hardware
//
#include <hip/hip_runtime.h>
#include <math.h>

// ---------------- problem constants ----------------
#define LNUM 12
#define DDIM 768
#define IDIM 3072
#define HNUM 12
#define DHEAD 64
#define BNUM 32
#define SLEN 197
#define MTOK (BNUM * SLEN)   // 6304 tokens
#define SPAD 224             // padded score row length (multiple of 32)
#define EPSLN 1e-5f
#define ATT_SCALE 0.125f     // 1/sqrt(64)

typedef __attribute__((ext_vector_type(16))) _Float16 v16h;
typedef __attribute__((ext_vector_type(8)))  _Float16 v8h;
typedef __attribute__((ext_vector_type(8)))  float    v8f;

// ======== weight repack: fp32 (KxN / head-blocked) -> f16 N-major [l][n][k] ====
__global__ __launch_bounds__(256)
void repack_kernel(const float* __restrict__ src, _Float16* __restrict__ dst,
                   int K, int N, int headBlocked, long long total) {
  long long per = (long long)K * N;
  for (long long i = (long long)blockIdx.x * blockDim.x + threadIdx.x; i < total;
       i += (long long)gridDim.x * blockDim.x) {
    long long l = i / per;
    long long r = i - l * per;
    int n = (int)(r / K);
    int k = (int)(r - (long long)n * K);
    float v;
    if (headBlocked)   // src (H, K, 64): n = h*64+e
      v = src[l * per + (long long)(n >> 6) * K * 64 + (long long)k * 64 + (n & 63)];
    else               // src row-major KxN
      v = src[l * per + (long long)k * N + n];
    dst[i] = (_Float16)v;
  }
}

// ======== LayerNorm: one block (256 thr) per token, f32 in -> f16 out ========
__global__ __launch_bounds__(256)
void ln_kernel(const float* __restrict__ x, const float* __restrict__ g,
               const float* __restrict__ b, _Float16* __restrict__ o) {
  int t = blockIdx.x;
  const float* xr = x + (size_t)t * DDIM;
  _Float16* orow = o + (size_t)t * DDIM;
  __shared__ float red[256];
  int tid = threadIdx.x;
  float v0 = xr[tid], v1 = xr[tid + 256], v2 = xr[tid + 512];
  red[tid] = v0 + v1 + v2;
  __syncthreads();
  #pragma unroll
  for (int off = 128; off > 0; off >>= 1) {
    if (tid < off) red[tid] += red[tid + off];
    __syncthreads();
  }
  float mean = red[0] * (1.0f / DDIM);
  __syncthreads();
  float d0 = v0 - mean, d1 = v1 - mean, d2 = v2 - mean;
  red[tid] = d0 * d0 + d1 * d1 + d2 * d2;
  __syncthreads();
  #pragma unroll
  for (int off = 128; off > 0; off >>= 1) {
    if (tid < off) red[tid] += red[tid + off];
    __syncthreads();
  }
  float r = rsqrtf(red[0] * (1.0f / DDIM) + EPSLN);
  orow[tid]       = (_Float16)(d0 * r * g[tid]       + b[tid]);
  orow[tid + 256] = (_Float16)(d1 * r * g[tid + 256] + b[tid + 256]);
  orow[tid + 512] = (_Float16)(d2 * r * g[tid + 512] + b[tid + 512]);
}

// ======== softmax: one wave32 per row; f32 scores -> f16 probs (224 padded) ===
__global__ __launch_bounds__(256)
void softmax_kernel(const float* __restrict__ s, _Float16* __restrict__ p, int nrows) {
  int row = blockIdx.x * 8 + (threadIdx.x >> 5);
  int lane = threadIdx.x & 31;
  if (row >= nrows) return;
  const float* sr = s + (size_t)row * SPAD;
  _Float16* pr = p + (size_t)row * SPAD;
  float vals[7];
  float mx = -3.4e38f;
  #pragma unroll
  for (int i = 0; i < 7; ++i) {
    int j = lane + i * 32;
    float v = (j < SLEN) ? sr[j] : -3.4e38f;
    vals[i] = v;
    mx = fmaxf(mx, v);
  }
  #pragma unroll
  for (int m = 16; m > 0; m >>= 1) mx = fmaxf(mx, __shfl_xor(mx, m, 32));
  float sum = 0.0f;
  #pragma unroll
  for (int i = 0; i < 7; ++i) {
    int j = lane + i * 32;
    float e = (j < SLEN) ? __expf(vals[i] - mx) : 0.0f;
    vals[i] = e;
    sum += e;
  }
  #pragma unroll
  for (int m = 16; m > 0; m >>= 1) sum += __shfl_xor(sum, m, 32);
  float inv = 1.0f / sum;
  #pragma unroll
  for (int i = 0; i < 7; ++i)
    pr[lane + i * 32] = (_Float16)(vals[i] * inv);   // 7*32 == 224, pad = zeros
}

// ======== f16 WMMA GEMM: double-buffered async global->LDS pipeline ========
// C = act(alpha * A(f16,[M,K]) * B(f16, N-major [N,K])^T + bias + Res)
// OUTMODE: 0 = f32 store, 1 = f16 store, 2 = f16 transposed store into
//          vT[b][h][e][SPAD] (for the V projection).
#define BM 128
#define BN 128
#define BK 32
#define LDF 40   // 80B row stride: 16B-aligned chunks, conflict-free b128 reads

__device__ __forceinline__ void issue_group(unsigned la0, const _Float16* pa0,
                                            unsigned la1, const _Float16* pa1,
                                            unsigned lb0, const _Float16* pb0,
                                            unsigned lb1, const _Float16* pb1) {
  // CDNA5 async copies: global -> LDS, no VGPR round trip (ASYNCcnt tracked)
  asm volatile("global_load_async_to_lds_b128 %0, %1, off"
               :: "v"(la0), "v"(pa0) : "memory");
  asm volatile("global_load_async_to_lds_b128 %0, %1, off"
               :: "v"(la1), "v"(pa1) : "memory");
  asm volatile("global_load_async_to_lds_b128 %0, %1, off"
               :: "v"(lb0), "v"(pb0) : "memory");
  asm volatile("global_load_async_to_lds_b128 %0, %1, off"
               :: "v"(lb1), "v"(pb1) : "memory");
}

__device__ __forceinline__ void wait_async(bool keep4) {
  if (keep4)  // previous group done; the 4 just-issued may stay in flight
    asm volatile("s_wait_asynccnt 0x4" ::: "memory");
  else
    asm volatile("s_wait_asynccnt 0x0" ::: "memory");
}

__device__ __forceinline__ void compute_step(const _Float16 (*Asb)[LDF],
                                             const _Float16 (*Bsb)[LDF],
                                             int waveM, int waveN, int l16, int lhalf,
                                             v8f (&acc)[2][4]) {
  v16h af[2], bf[4];
  #pragma unroll
  for (int fi = 0; fi < 2; ++fi) {
    const _Float16* ap = &Asb[waveM * 32 + fi * 16 + l16][lhalf * 16];
    v8h alo = *(const v8h*)ap;
    v8h ahi = *(const v8h*)(ap + 8);
    af[fi] = __builtin_shufflevector(alo, ahi,
             0, 1, 2, 3, 4, 5, 6, 7, 8, 9, 10, 11, 12, 13, 14, 15);
  }
  #pragma unroll
  for (int fj = 0; fj < 4; ++fj) {
    const _Float16* bp = &Bsb[waveN * 64 + fj * 16 + l16][lhalf * 16];
    v8h blo = *(const v8h*)bp;
    v8h bhi = *(const v8h*)(bp + 8);
    bf[fj] = __builtin_shufflevector(blo, bhi,
             0, 1, 2, 3, 4, 5, 6, 7, 8, 9, 10, 11, 12, 13, 14, 15);
  }
  #pragma unroll
  for (int fi = 0; fi < 2; ++fi)
    #pragma unroll
    for (int fj = 0; fj < 4; ++fj)
      acc[fi][fj] = __builtin_amdgcn_wmma_f32_16x16x32_f16(
          false, af[fi], false, bf[fj], (short)0, acc[fi][fj], false, false);
}

template <int OUTMODE>
__global__ __launch_bounds__(256)
void gemm_f16_wmma(const _Float16* __restrict__ A, int lda, long long sAb, long long sAh,
                   const _Float16* __restrict__ B, int ldb, long long sBb, long long sBh,
                   const float* __restrict__ bias, const float* __restrict__ Res, int ldr,
                   void* __restrict__ Cv, int ldc, long long sCb, long long sCh,
                   int M, int N, int K, int nheads,
                   float alpha, int doGelu) {
  __shared__ __align__(16) _Float16 As[2][BM][LDF];
  __shared__ __align__(16) _Float16 Bs[2][BN][LDF];

  int z = blockIdx.z;
  int bb = z / nheads, hh = z % nheads;
  const _Float16* Ab = A + bb * sAb + hh * sAh;
  const _Float16* Bb = B + bb * sBb + hh * sBh;

  int m0 = blockIdx.y * BM;
  int n0 = blockIdx.x * BN;

  int tid = threadIdx.x;
  int lane = tid & 31;
  int wave = tid >> 5;
  int waveM = wave & 3;    // 4 waves along M (32 rows each)
  int waveN = wave >> 2;   // 2 waves along N (64 cols each)
  int l16 = lane & 15;
  int lhalf = lane >> 4;

  // staging: 512 chunks of 16B per operand; each thread moves 2 A + 2 B chunks.
  int c0 = tid, c1 = tid + 256;
  int ar0 = c0 >> 2, ak0 = (c0 & 3) * 8;
  int ar1 = c1 >> 2, ak1 = (c1 & 3) * 8;
  // clamp out-of-range rows: garbage lands only in discarded output rows/cols
  long long gmA0 = min(m0 + ar0, M - 1);
  long long gmA1 = min(m0 + ar1, M - 1);
  long long gnB0 = min(n0 + ar0, N - 1);
  long long gnB1 = min(n0 + ar1, N - 1);

  const _Float16* paB0 = Ab + gmA0 * lda + ak0;
  const _Float16* paB1 = Ab + gmA1 * lda + ak1;
  const _Float16* pbB0 = Bb + gnB0 * ldb + ak0;
  const _Float16* pbB1 = Bb + gnB1 * ldb + ak1;

  unsigned lA0[2], lA1[2], lB0[2], lB1[2];
  #pragma unroll
  for (int u = 0; u < 2; ++u) {
    lA0[u] = (unsigned)(unsigned long long)&As[u][ar0][ak0];
    lA1[u] = (unsigned)(unsigned long long)&As[u][ar1][ak1];
    lB0[u] = (unsigned)(unsigned long long)&Bs[u][ar0][ak0];
    lB1[u] = (unsigned)(unsigned long long)&Bs[u][ar1][ak1];
  }

  v8f acc[2][4] = {};

  int ktiles = K / BK;   // all K are multiples of 32
  // prologue: stage tile 0 into buffer 0
  issue_group(lA0[0], paB0, lA1[0], paB1, lB0[0], pbB0, lB1[0], pbB1);

  int kt = 0;
  while (kt < ktiles) {
    {   // ---- compute from buffer 0, prefetch kt+1 into buffer 1 ----
      bool hn = (kt + 1) < ktiles;
      if (hn) {
        int k1 = (kt + 1) * BK;
        issue_group(lA0[1], paB0 + k1, lA1[1], paB1 + k1,
                    lB0[1], pbB0 + k1, lB1[1], pbB1 + k1);
      }
      wait_async(hn);
      __syncthreads();
      compute_step(As[0], Bs[0], waveM, waveN, l16, lhalf, acc);
      __syncthreads();
      ++kt;
      if (kt >= ktiles) break;
    }
    {   // ---- compute from buffer 1, prefetch kt+1 into buffer 0 ----
      bool hn = (kt + 1) < ktiles;
      if (hn) {
        int k1 = (kt + 1) * BK;
        issue_group(lA0[0], paB0 + k1, lA1[0], paB1 + k1,
                    lB0[0], pbB0 + k1, lB1[0], pbB1 + k1);
      }
      wait_async(hn);
      __syncthreads();
      compute_step(As[1], Bs[1], waveM, waveN, l16, lhalf, acc);
      __syncthreads();
      ++kt;
    }
  }

  // ---- epilogue (dead-code-stripped per OUTMODE) ----
  #pragma unroll
  for (int fi = 0; fi < 2; ++fi) {
    #pragma unroll
    for (int fj = 0; fj < 4; ++fj) {
      int mbase = m0 + waveM * 32 + fi * 16 + lhalf * 8;
      int nn = n0 + waveN * 64 + fj * 16 + l16;
      #pragma unroll
      for (int r = 0; r < 8; ++r) {
        int gm = mbase + r;
        if (gm < M && nn < N) {
          float v = acc[fi][fj][r] * alpha;
          if (bias) v += bias[nn];
          if (Res)  v += Res[(long long)gm * ldr + nn];
          if (doGelu) v = 0.5f * v * (1.0f + erff(v * 0.70710678118f));
          if (OUTMODE == 0) {
            float* Cf = (float*)Cv + bb * sCb + hh * sCh;
            Cf[(long long)gm * ldc + nn] = v;
          } else if (OUTMODE == 1) {
            _Float16* Ch = (_Float16*)Cv + bb * sCb + hh * sCh;
            Ch[(long long)gm * ldc + nn] = (_Float16)v;
          } else {  // V projection -> vT[b][h][e][SPAD]
            _Float16* Ch = (_Float16*)Cv;
            int bidx = gm / SLEN;
            int t = gm - bidx * SLEN;
            int h = nn >> 6, e = nn & 63;
            Ch[(((long long)bidx * HNUM + h) * DHEAD + e) * SPAD + t] = (_Float16)v;
          }
        }
      }
    }
  }
}

// ---------------- host orchestration ----------------
static inline dim3 gemm_grid(int M, int N, int batches) {
  return dim3((N + BN - 1) / BN, (M + BM - 1) / BM, batches);
}

extern "C" void kernel_launch(void* const* d_in, const int* in_sizes, int n_in,
                              void* d_out, int out_size, void* d_ws, size_t ws_size,
                              hipStream_t stream) {
  const float* X   = (const float*)d_in[0];
  const float* Wq  = (const float*)d_in[1];
  const float* bq  = (const float*)d_in[2];
  const float* Wk  = (const float*)d_in[3];
  const float* bk  = (const float*)d_in[4];
  const float* Wv  = (const float*)d_in[5];
  const float* bv  = (const float*)d_in[6];
  const float* Wo  = (const float*)d_in[7];
  const float* bo  = (const float*)d_in[8];
  const float* W1  = (const float*)d_in[9];
  const float* b1  = (const float*)d_in[10];
  const float* W2  = (const float*)d_in[11];
  const float* b2  = (const float*)d_in[12];
  const float* g1  = (const float*)d_in[13];
  const float* be1 = (const float*)d_in[14];
  const float* g2  = (const float*)d_in[15];
  const float* be2 = (const float*)d_in[16];

  // ---- carve workspace (16B aligned slabs) ----
  char* base = (char*)d_ws;
  size_t off = 0;
  auto carve = [&](size_t bytes) {
    char* p = base + off;
    off += (bytes + 15) & ~(size_t)15;
    return p;
  };
  float*    xA    = (float*)   carve((size_t)MTOK * DDIM * 4);
  float*    xB    = (float*)   carve((size_t)MTOK * DDIM * 4);
  float*    scor  = (float*)   carve((size_t)BNUM * HNUM * SLEN * SPAD * 4);
  _Float16* wq_h  = (_Float16*)carve((size_t)LNUM * DDIM * DDIM * 2);
  _Float16* wk_h  = (_Float16*)carve((size_t)LNUM * DDIM * DDIM * 2);
  _Float16* wv_h  = (_Float16*)carve((size_t)LNUM * DDIM * DDIM * 2);
  _Float16* wo_h  = (_Float16*)carve((size_t)LNUM * DDIM * DDIM * 2);
  _Float16* w1_h  = (_Float16*)carve((size_t)LNUM * DDIM * IDIM * 2);
  _Float16* w2_h  = (_Float16*)carve((size_t)LNUM * DDIM * IDIM * 2);
  _Float16* xn_h  = (_Float16*)carve((size_t)MTOK * DDIM * 2);
  _Float16* q_h   = (_Float16*)carve((size_t)MTOK * DDIM * 2);
  _Float16* k_h   = (_Float16*)carve((size_t)MTOK * DDIM * 2);
  _Float16* vT_h  = (_Float16*)carve((size_t)BNUM * HNUM * DHEAD * SPAD * 2);
  _Float16* at_h  = (_Float16*)carve((size_t)MTOK * DDIM * 2);
  _Float16* h_h   = (_Float16*)carve((size_t)MTOK * IDIM * 2);
  _Float16* pr_h  = (_Float16*)carve((size_t)BNUM * HNUM * SLEN * SPAD * 2);

  // ---- one-time (per call) weight convert + repack to f16 N-major ----
  {
    long long tq = (long long)LNUM * DDIM * DDIM;
    long long t1 = (long long)LNUM * DDIM * IDIM;
    repack_kernel<<<2048, 256, 0, stream>>>(Wq, wq_h, DDIM, DDIM, 1, tq);
    repack_kernel<<<2048, 256, 0, stream>>>(Wk, wk_h, DDIM, DDIM, 1, tq);
    repack_kernel<<<2048, 256, 0, stream>>>(Wv, wv_h, DDIM, DDIM, 1, tq);
    repack_kernel<<<2048, 256, 0, stream>>>(Wo, wo_h, DDIM, DDIM, 0, tq);
    repack_kernel<<<2048, 256, 0, stream>>>(W1, w1_h, DDIM, IDIM, 0, t1);
    repack_kernel<<<2048, 256, 0, stream>>>(W2, w2_h, IDIM, DDIM, 0, t1);
  }

  hipMemcpyAsync(xA, X, (size_t)MTOK * DDIM * sizeof(float),
                 hipMemcpyDeviceToDevice, stream);

  const long long sQb = (long long)SLEN * DDIM;        // (b) stride in [b,s,h*64+e]
  const long long sSb = (long long)HNUM * SLEN * SPAD; // (b) stride in scores/probs
  const long long sSh = (long long)SLEN * SPAD;        // (h) stride
  const long long sVz = (long long)DHEAD * SPAD;       // per-(b,h) stride in vT

  for (int l = 0; l < LNUM; ++l) {
    _Float16* wq = wq_h + (size_t)l * DDIM * DDIM;
    _Float16* wk = wk_h + (size_t)l * DDIM * DDIM;
    _Float16* wv = wv_h + (size_t)l * DDIM * DDIM;
    _Float16* wo = wo_h + (size_t)l * DDIM * DDIM;
    _Float16* w1 = w1_h + (size_t)l * DDIM * IDIM;
    _Float16* w2 = w2_h + (size_t)l * IDIM * DDIM;
    const float* bql = bq + (size_t)l * DDIM;
    const float* bkl = bk + (size_t)l * DDIM;
    const float* bvl = bv + (size_t)l * DDIM;
    const float* bol = bo + (size_t)l * DDIM;
    const float* b1l = b1 + (size_t)l * IDIM;
    const float* b2l = b2 + (size_t)l * DDIM;

    // LN1: f32 -> f16
    ln_kernel<<<MTOK, 256, 0, stream>>>(xA, g1 + (size_t)l * DDIM,
                                        be1 + (size_t)l * DDIM, xn_h);
    // Q = xn Wq + bq  (f16 out)
    gemm_f16_wmma<1><<<gemm_grid(MTOK, DDIM, 1), 256, 0, stream>>>(
        xn_h, DDIM, 0, 0, wq, DDIM, 0, 0, bql, nullptr, 0,
        q_h, DDIM, 0, 0, MTOK, DDIM, DDIM, 1, 1.0f, 0);
    // K
    gemm_f16_wmma<1><<<gemm_grid(MTOK, DDIM, 1), 256, 0, stream>>>(
        xn_h, DDIM, 0, 0, wk, DDIM, 0, 0, bkl, nullptr, 0,
        k_h, DDIM, 0, 0, MTOK, DDIM, DDIM, 1, 1.0f, 0);
    // V -> transposed f16 vT[b][h][e][224]
    gemm_f16_wmma<2><<<gemm_grid(MTOK, DDIM, 1), 256, 0, stream>>>(
        xn_h, DDIM, 0, 0, wv, DDIM, 0, 0, bvl, nullptr, 0,
        vT_h, 0, 0, 0, MTOK, DDIM, DDIM, 1, 1.0f, 0);
    // scores = SCALE * Q K^T  (batched; K matrix is already N-major per (b,h))
    gemm_f16_wmma<0><<<gemm_grid(SLEN, SLEN, BNUM * HNUM), 256, 0, stream>>>(
        q_h, DDIM, sQb, DHEAD, k_h, DDIM, sQb, DHEAD, nullptr, nullptr, 0,
        scor, SPAD, sSb, sSh, SLEN, SLEN, DHEAD, HNUM, ATT_SCALE, 0);
    // softmax -> f16 probs, zero pad to 224
    {
      int nrows = BNUM * HNUM * SLEN;
      softmax_kernel<<<(nrows + 7) / 8, 256, 0, stream>>>(scor, pr_h, nrows);
    }
    // attn = probs @ V  (B = vT, N-major; pad cols hit zero probs)
    gemm_f16_wmma<1><<<gemm_grid(SLEN, DHEAD, BNUM * HNUM), 256, 0, stream>>>(
        pr_h, SPAD, sSb, sSh, vT_h, SPAD, (long long)HNUM * sVz, sVz,
        nullptr, nullptr, 0,
        at_h, DDIM, sQb, DHEAD, SLEN, DHEAD, SPAD, HNUM, 1.0f, 0);
    // x = x + attn Wo + bo  (f32 out)
    gemm_f16_wmma<0><<<gemm_grid(MTOK, DDIM, 1), 256, 0, stream>>>(
        at_h, DDIM, 0, 0, wo, DDIM, 0, 0, bol, xA, DDIM,
        xB, DDIM, 0, 0, MTOK, DDIM, DDIM, 1, 1.0f, 0);
    // LN2
    ln_kernel<<<MTOK, 256, 0, stream>>>(xB, g2 + (size_t)l * DDIM,
                                        be2 + (size_t)l * DDIM, xn_h);
    // h = gelu(xn W1 + b1)  (f16 out)
    gemm_f16_wmma<1><<<gemm_grid(MTOK, IDIM, 1), 256, 0, stream>>>(
        xn_h, DDIM, 0, 0, w1, DDIM, 0, 0, b1l, nullptr, 0,
        h_h, IDIM, 0, 0, MTOK, IDIM, DDIM, 1, 1.0f, 1);
    // x = x + h W2 + b2  (f32 out)
    gemm_f16_wmma<0><<<gemm_grid(MTOK, DDIM, 1), 256, 0, stream>>>(
        h_h, IDIM, 0, 0, w2, IDIM, 0, 0, b2l, xB, DDIM,
        xA, DDIM, 0, 0, MTOK, DDIM, IDIM, 1, 1.0f, 0);
  }

  hipMemcpyAsync(d_out, xA, (size_t)MTOK * DDIM * sizeof(float),
                 hipMemcpyDeviceToDevice, stream);
}
